// GhostTTS_40518721471153
// MI455X (gfx1250) — compile-verified, matrix operand order checked by reading, not run on hardware
//
#include <hip/hip_runtime.h>

// ---------------- model constants (from reference) ----------------
#define DV   512     // D
#define HN   8       // heads
#define HD   64      // head dim
#define LN_  4       // layers
#define BN_  4       // batch
#define TN_  512     // text length
#define MLN  1024    // max mel frames
#define FFN  2048    // FF dim
#define FSN  256     // varpred filter size
#define NMELC 80     // mel channels
#define PNC  512     // postnet channels

typedef _Float16 f16;
typedef __attribute__((ext_vector_type(16))) _Float16 v16h;
typedef __attribute__((ext_vector_type(8)))  float    v8f;

#define EPI_BIAS  1
#define EPI_ACCUM 2
#define EPI_GELU  4

// =================================================================
// WMMA f16 GEMM:  C[M,N] (f32) = A[M,K] (f16) @ B[N,K]^T (f16)
// Block tile 128x128, 8 waves in 2(M) x 4(N); wave tile 64x32 =
// 4x2 WMMA 16x16x32 tiles (8 v_wmma per K-step). Global->LDS staging
// is register-pipelined; per-row remap/validity is resolved ONCE
// before the K-loop. Invalid rows are CLAMPED to row 0 (addresses
// always in-bounds, loads unconditional -> global_load_b128, no
// flat fallback); invalid data is zeroed with loop-invariant masks.
// Requires K % 8 == 0 (true for all GEMMs in this model).
// Optional per-batch row shift on A (segLen>0): row m ->
// b*segLen + (m%segLen + shift), zero outside segment (conv taps).
// Epilogue: bias / accumulate / exact GELU, optional f16 mirror C16.
// =================================================================
__device__ __forceinline__ v16h frag_ld(const f16* p)
{
    union { uint4 u[2]; v16h v; } f;
    f.u[0] = *(const uint4*)p;        // k = kb .. kb+7
    f.u[1] = *(const uint4*)(p + 16); // k = kb+16 .. kb+23
    return f.v;
}

__global__ void __launch_bounds__(256)
gemm_f16_wmma(const f16* __restrict__ A, const f16* __restrict__ Bw,
              float* __restrict__ C, const float* __restrict__ bias,
              f16* __restrict__ C16,
              int M, int N, int K, int flags, int segLen, int shift)
{
    __shared__ __align__(16) f16 lA[128 * 40];
    __shared__ __align__(16) f16 lB[128 * 40];

    const int tid  = threadIdx.x;
    const int lane = tid & 31;
    const int wid  = tid >> 5;
    const int wm   = (wid & 1) << 6;   // 0 / 64
    const int wn   = (wid >> 1) << 5;  // 0 / 32 / 64 / 96
    const int m0   = blockIdx.y * 128;
    const int n0   = blockIdx.x * 128;

    // staging chunk coords: rows r0 and r0+64, 8 halfs starting at c80
    const int r0  = tid >> 2;
    const int c80 = (tid & 3) << 3;

    // ---- resolve staging rows once (remap + validity), loop-invariant ----
    // invalid rows are clamped to row 0 so every address stays in-bounds
    bool okA0, okA1, okB0, okB1;
    const f16* pA0; const f16* pA1; const f16* pB0; const f16* pB1;
    {
        int gm0 = m0 + r0, gm1 = m0 + r0 + 64;
        long s0 = gm0, s1 = gm1;
        okA0 = gm0 < M; okA1 = gm1 < M;
        if (segLen > 0) {
            if (okA0) {
                int bb = gm0 / segLen, ff = gm0 - bb * segLen + shift;
                okA0 = (ff >= 0) && (ff < segLen);
                s0 = (long)bb * segLen + ff;
            }
            if (okA1) {
                int bb = gm1 / segLen, ff = gm1 - bb * segLen + shift;
                okA1 = (ff >= 0) && (ff < segLen);
                s1 = (long)bb * segLen + ff;
            }
        }
        pA0 = A + (okA0 ? s0 * (long)K : 0) + c80;
        pA1 = A + (okA1 ? s1 * (long)K : 0) + c80;
        int gn0 = n0 + r0, gn1 = n0 + r0 + 64;
        okB0 = gn0 < N; okB1 = gn1 < N;
        pB0 = Bw + (okB0 ? (long)gn0 * K : 0) + c80;
        pB1 = Bw + (okB1 ? (long)gn1 * K : 0) + c80;
    }

    const uint4 z4 = {0u, 0u, 0u, 0u};
    // unconditional loads from clamped addresses; zero data via invariant masks
    auto fetch = [&](int kk, uint4& a0, uint4& a1, uint4& b0, uint4& b1) {
        const bool kin = (kk + c80 + 8) <= K;   // K % 8 == 0 -> exact
        const int  kkc = kin ? kk : 0;          // keep address in-bounds
        a0 = *(const uint4*)(pA0 + kkc);
        a1 = *(const uint4*)(pA1 + kkc);
        b0 = *(const uint4*)(pB0 + kkc);
        b1 = *(const uint4*)(pB1 + kkc);
        if (!(okA0 && kin)) a0 = z4;
        if (!(okA1 && kin)) a1 = z4;
        if (!(okB0 && kin)) b0 = z4;
        if (!(okB1 && kin)) b1 = z4;
    };

    v8f acc[4][2];
    for (int i = 0; i < 4; ++i)
        for (int j = 0; j < 2; ++j)
            for (int e = 0; e < 8; ++e) acc[i][j][e] = 0.0f;

    // prologue: fetch first K-tile into registers
    uint4 ra0, ra1, rb0, rb1;
    fetch(0, ra0, ra1, rb0, rb1);

    for (int k0 = 0; k0 < K; k0 += 32) {
        // commit current K-tile registers to LDS
        *(uint4*)&lA[r0 * 40 + c80]        = ra0;
        *(uint4*)&lA[(r0 + 64) * 40 + c80] = ra1;
        *(uint4*)&lB[r0 * 40 + c80]        = rb0;
        *(uint4*)&lB[(r0 + 64) * 40 + c80] = rb1;
        __syncthreads();

        // pipeline: fetch next K-tile while WMMAs run from LDS
        const int kn = k0 + 32;
        if (kn < K) {
            fetch(kn, ra0, ra1, rb0, rb1);
            // hint the tile after next into cache (global_prefetch_b8)
            if (kn + 32 < K) {
                if (okA0) __builtin_prefetch(pA0 + kn + 32, 0, 1);
                if (okB0) __builtin_prefetch(pB0 + kn + 32, 0, 1);
            }
        }

        // fragment loads (16-bit A 16x32 lane layout per ISA 7.12.2)
        const int rA = lane & 15;
        const int kb = (lane < 16) ? 0 : 8;
        v16h af[4], bfm[2];
        for (int t = 0; t < 4; ++t)
            af[t] = frag_ld(&lA[(wm + t * 16 + rA) * 40 + kb]);
        for (int t = 0; t < 2; ++t)
            bfm[t] = frag_ld(&lB[(wn + t * 16 + rA) * 40 + kb]);

        for (int tm = 0; tm < 4; ++tm)
            for (int tn = 0; tn < 2; ++tn)
                acc[tm][tn] = __builtin_amdgcn_wmma_f32_16x16x32_f16(
                    false, af[tm], false, bfm[tn], (short)0, acc[tm][tn],
                    false, false);
        __syncthreads();
    }

    // epilogue: 16x16 f32 C layout (VGPR r -> row base+r / base+8+r)
    for (int tm = 0; tm < 4; ++tm) {
        for (int tn = 0; tn < 2; ++tn) {
            int col = n0 + wn + tn * 16 + (lane & 15);
            if (col >= N) continue;
            int rowb = m0 + wm + tm * 16 + ((lane < 16) ? 0 : 8);
            for (int r = 0; r < 8; ++r) {
                int row = rowb + r;
                if (row >= M) break;
                float v = acc[tm][tn][r];
                long off = (long)row * N + col;
                if (flags & EPI_ACCUM) v += C[off];
                if (flags & EPI_BIAS)  v += bias[col];
                if (flags & EPI_GELU)  v = 0.5f * v * (1.0f + erff(v * 0.70710678118f));
                C[off] = v;
                if (C16) C16[off] = (f16)v;
            }
        }
    }
}

// ================= dequant / conversion kernels =================
__global__ void dequant_k(const int* __restrict__ idx, const float* __restrict__ scale,
                          const float* __restrict__ lut, f16* __restrict__ out,
                          long O, long K)
{
    long n = O * K;
    for (long i = (long)blockIdx.x * blockDim.x + threadIdx.x; i < n;
         i += (long)gridDim.x * blockDim.x)
        out[i] = (f16)(lut[idx[i]] * scale[i / K]);
}

// conv weight [O,I,Kt] f32 -> taps [Kt][O][I] f16
__global__ void taps_k(const float* __restrict__ w, f16* __restrict__ out,
                       long O, long I, long Kt)
{
    long n = O * I * Kt;
    for (long i = (long)blockIdx.x * blockDim.x + threadIdx.x; i < n;
         i += (long)gridDim.x * blockDim.x) {
        long kt = i % Kt;
        long oi = i / Kt;
        out[kt * O * I + oi] = (f16)w[i];
    }
}

__global__ void cvt_f16_k(const float* __restrict__ in, f16* __restrict__ out, long n)
{
    for (long i = (long)blockIdx.x * blockDim.x + threadIdx.x; i < n;
         i += (long)gridDim.x * blockDim.x)
        out[i] = (f16)in[i];
}

// ================= small op kernels =================
__global__ void embed_k(const int* __restrict__ text, const float* __restrict__ emb,
                        const float* __restrict__ pos, float* __restrict__ x,
                        int Tt, int C, long n)
{
    for (long i = (long)blockIdx.x * blockDim.x + threadIdx.x; i < n;
         i += (long)gridDim.x * blockDim.x) {
        int c   = (int)(i % C);
        long bt = i / C;
        int t   = (int)(bt % Tt);
        x[i] = emb[(long)text[bt] * C + c] + pos[(long)t * C + c];
    }
}

__global__ void __launch_bounds__(256)
layernorm_k(const float* __restrict__ x, float* __restrict__ y,
            const float* __restrict__ g, const float* __restrict__ b, int C)
{
    __shared__ float red[256];
    const int tid = threadIdx.x;
    const long base = (long)blockIdx.x * C;
    float s = 0.f;
    for (int c = tid; c < C; c += 256) s += x[base + c];
    red[tid] = s; __syncthreads();
    for (int s2 = 128; s2 > 0; s2 >>= 1) {
        if (tid < s2) red[tid] += red[tid + s2];
        __syncthreads();
    }
    float mean = red[0] / C; __syncthreads();
    float q = 0.f;
    for (int c = tid; c < C; c += 256) { float d = x[base + c] - mean; q += d * d; }
    red[tid] = q; __syncthreads();
    for (int s2 = 128; s2 > 0; s2 >>= 1) {
        if (tid < s2) red[tid] += red[tid + s2];
        __syncthreads();
    }
    float rstd = rsqrtf(red[0] / C + 1e-5f);
    for (int c = tid; c < C; c += 256)
        y[base + c] = (x[base + c] - mean) * rstd * g[c] + b[c];
}

// softmax attention: one block per (b,h,q).
// q/k/v share row stride rs (fused QKV buffer); out has row stride os.
__global__ void __launch_bounds__(256)
attention_k(const float* __restrict__ q, const float* __restrict__ k,
            const float* __restrict__ v, float* __restrict__ o,
            int Tt, int rs, int os)
{
    __shared__ float sc[MLN];
    __shared__ float qs[HD];
    __shared__ float red[256];
    const int tid = threadIdx.x;
    int bid = blockIdx.x;
    int t = bid % Tt; bid /= Tt;
    int h = bid % HN;
    int b = bid / HN;
    const float* qp = q + ((long)(b * Tt + t) * rs) + h * HD;
    if (tid < HD) qs[tid] = qp[tid];
    __syncthreads();
    for (int kk = tid; kk < Tt; kk += 256) {
        const float* kp = k + ((long)(b * Tt + kk) * rs) + h * HD;
        float s = 0.f;
        #pragma unroll 8
        for (int d = 0; d < HD; ++d) s += qs[d] * kp[d];
        sc[kk] = s * 0.125f;   // 1/sqrt(64)
    }
    __syncthreads();
    float m = -3.0e38f;
    for (int kk = tid; kk < Tt; kk += 256) m = fmaxf(m, sc[kk]);
    red[tid] = m; __syncthreads();
    for (int s2 = 128; s2 > 0; s2 >>= 1) {
        if (tid < s2) red[tid] = fmaxf(red[tid], red[tid + s2]);
        __syncthreads();
    }
    m = red[0]; __syncthreads();
    float zs = 0.f;
    for (int kk = tid; kk < Tt; kk += 256) {
        float p = __expf(sc[kk] - m);
        sc[kk] = p;
        zs += p;
    }
    red[tid] = zs; __syncthreads();
    for (int s2 = 128; s2 > 0; s2 >>= 1) {
        if (tid < s2) red[tid] += red[tid + s2];
        __syncthreads();
    }
    float Z = red[0];
    __syncthreads();
    if (tid < HD) {
        float a = 0.f;
        for (int kk = 0; kk < Tt; ++kk)
            a += sc[kk] * v[((long)(b * Tt + kk) * rs) + h * HD + tid];
        o[((long)(b * Tt + t) * os) + h * HD + tid] = a / Z;
    }
}

__global__ void bias_relu_k(float* __restrict__ x, const float* __restrict__ bias,
                            long n, int C)
{
    for (long i = (long)blockIdx.x * blockDim.x + threadIdx.x; i < n;
         i += (long)gridDim.x * blockDim.x) {
        float v = x[i] + bias[i % C];
        x[i] = v > 0.f ? v : 0.f;
    }
}

// BN(eval) + tanh, also emits f16 mirror for next WMMA layer
__global__ void bn_tanh_k(float* __restrict__ x, const float* __restrict__ bias,
                          const float* __restrict__ g, const float* __restrict__ bb,
                          f16* __restrict__ out16, long n, int C)
{
    const float inv = 0.9999950000374997f; // 1/sqrt(1+1e-5)
    for (long i = (long)blockIdx.x * blockDim.x + threadIdx.x; i < n;
         i += (long)gridDim.x * blockDim.x) {
        int c = (int)(i % C);
        float v = x[i] + bias[c];
        v = tanhf(v * inv * g[c] + bb[c]);
        x[i] = v;
        out16[i] = (f16)v;
    }
}

__global__ void __launch_bounds__(256)
rowdot_k(const float* __restrict__ h, const float* __restrict__ w,
         const float* __restrict__ b, float* __restrict__ out, int C)
{
    __shared__ float red[256];
    const int tid = threadIdx.x;
    long base = (long)blockIdx.x * C;
    float s = 0.f;
    for (int c = tid; c < C; c += 256) s += h[base + c] * w[c];
    red[tid] = s; __syncthreads();
    for (int s2 = 128; s2 > 0; s2 >>= 1) {
        if (tid < s2) red[tid] += red[tid + s2];
        __syncthreads();
    }
    if (tid == 0) out[blockIdx.x] = red[0] + b[0];
}

__global__ void cumsum_k(const int* __restrict__ dur, int* __restrict__ cum, int Tt)
{
    if (threadIdx.x == 0) {
        int b = blockIdx.x, c = 0;
        for (int t = 0; t < Tt; ++t) { c += dur[b * Tt + t]; cum[b * Tt + t] = c; }
    }
}

__global__ void __launch_bounds__(256)
lengthreg_k(const float* __restrict__ x, const int* __restrict__ cum,
            float* __restrict__ y, int Tt, int Ml, int C)
{
    int bid = blockIdx.x;
    int f = bid % Ml;
    int b = bid / Ml;
    int lo = 0, hi = Tt;
    while (lo < hi) {
        int mid = (lo + hi) >> 1;
        if (cum[b * Tt + mid] <= f) lo = mid + 1; else hi = mid;
    }
    int tok = lo < Tt - 1 ? lo : Tt - 1;
    float valid = (f < cum[b * Tt + Tt - 1]) ? 1.f : 0.f;
    const float* src = x + ((long)(b * Tt + tok) * C);
    float* dst = y + ((long)(b * Ml + f) * C);
    for (int c = threadIdx.x; c < C; c += 256) dst[c] = src[c] * valid;
}

// x[row,c] += s[row]*w[c] + b[c]
__global__ void addsc_k(float* __restrict__ x, const float* __restrict__ s,
                        const float* __restrict__ w, const float* __restrict__ b,
                        long n, int C)
{
    for (long i = (long)blockIdx.x * blockDim.x + threadIdx.x; i < n;
         i += (long)gridDim.x * blockDim.x) {
        int c = (int)(i % C);
        x[i] += s[i / C] * w[c] + b[c];
    }
}

__global__ void addpos_k(float* __restrict__ x, const float* __restrict__ pos,
                         long n, int Fr, int C)
{
    for (long i = (long)blockIdx.x * blockDim.x + threadIdx.x; i < n;
         i += (long)gridDim.x * blockDim.x) {
        int c = (int)(i % C);
        long row = i / C;
        int f = (int)(row % Fr);
        x[i] += pos[(long)f * C + c];
    }
}

__global__ void postadd_k(const float* __restrict__ mel, const float* __restrict__ t,
                          const float* __restrict__ b5, float* __restrict__ out,
                          long n, int C)
{
    for (long i = (long)blockIdx.x * blockDim.x + threadIdx.x; i < n;
         i += (long)gridDim.x * blockDim.x)
        out[i] = mel[i] + t[i] + b5[i % C];
}

// =================================================================
extern "C" void kernel_launch(void* const* d_in, const int* in_sizes, int n_in,
                              void* d_out, int out_size, void* d_ws, size_t ws_size,
                              hipStream_t stream)
{
    (void)in_sizes; (void)n_in; (void)out_size; (void)ws_size;

    // ---- inputs (setup_inputs dict order) ----
    const int*   text   = (const int*)  d_in[0];
    const int*   durt   = (const int*)  d_in[1];
    const float* lut    = (const float*)d_in[2];
    const float* emb    = (const float*)d_in[3];
    const float* pos    = (const float*)d_in[4];
    const float* dpos   = (const float*)d_in[5];
    const int*   e_ai   = (const int*)  d_in[6];
    const float* e_as   = (const float*)d_in[7];
    const int*   e_ui   = (const int*)  d_in[8];
    const float* e_us   = (const float*)d_in[9];
    const int*   e_di   = (const int*)  d_in[10];
    const float* e_ds   = (const float*)d_in[11];
    const float* e_g    = (const float*)d_in[12];
    const float* e_b    = (const float*)d_in[13];
    const int*   d_ai   = (const int*)  d_in[14];
    const float* d_as   = (const float*)d_in[15];
    const int*   d_ui   = (const int*)  d_in[16];
    const float* d_us   = (const float*)d_in[17];
    const int*   d_di   = (const int*)  d_in[18];
    const float* d_ds   = (const float*)d_in[19];
    const float* d_g    = (const float*)d_in[20];
    const float* d_b    = (const float*)d_in[21];
    const float* vp_c1w = (const float*)d_in[22];
    const float* vp_c1b = (const float*)d_in[23];
    const float* vp_g1  = (const float*)d_in[24];
    const float* vp_b1  = (const float*)d_in[25];
    const float* vp_c2w = (const float*)d_in[26];
    const float* vp_c2b = (const float*)d_in[27];
    const float* vp_g2  = (const float*)d_in[28];
    const float* vp_b2  = (const float*)d_in[29];
    const float* vp_lw  = (const float*)d_in[30];
    const float* vp_lb  = (const float*)d_in[31];
    const float* proj_w = (const float*)d_in[32];
    const float* proj_b = (const float*)d_in[33];
    const float* mel_w  = (const float*)d_in[34];
    const float* mel_b  = (const float*)d_in[35];
    const float* pn_w1  = (const float*)d_in[36];
    const float* pn_b1  = (const float*)d_in[37];
    const float* pn_w2  = (const float*)d_in[38];
    const float* pn_b2  = (const float*)d_in[39];
    const float* pn_w3  = (const float*)d_in[40];
    const float* pn_b3  = (const float*)d_in[41];
    const float* pn_w4  = (const float*)d_in[42];
    const float* pn_b4  = (const float*)d_in[43];
    const float* pn_w5  = (const float*)d_in[44];
    const float* pn_b5  = (const float*)d_in[45];
    const float* pn_g   = (const float*)d_in[46];
    const float* pn_bb  = (const float*)d_in[47];

    // ---- outputs ----
    float* out      = (float*)d_out;
    float* o_mel    = out;
    float* o_post   = o_mel  + (size_t)BN_ * MLN * NMELC;
    float* o_dur    = o_post + (size_t)BN_ * MLN * NMELC;
    float* o_pitch  = o_dur  + (size_t)BN_ * TN_;
    float* o_energy = o_pitch + (size_t)BN_ * MLN;

    // ---- workspace arena ----
    char* base = (char*)d_ws;
    size_t off = 0;
    auto alloc = [&](size_t bytes) -> void* {
        void* r = base + off;
        off = (off + bytes + 255) & ~(size_t)255;
        return r;
    };
    f16* WeA  = (f16*)alloc((size_t)LN_ * 4 * DV * DV * 2);
    f16* WeU  = (f16*)alloc((size_t)LN_ * FFN * DV * 2);
    f16* WeD  = (f16*)alloc((size_t)LN_ * DV * FFN * 2);
    f16* WdA  = (f16*)alloc((size_t)LN_ * 4 * DV * DV * 2);
    f16* WdU  = (f16*)alloc((size_t)LN_ * FFN * DV * 2);
    f16* WdD  = (f16*)alloc((size_t)LN_ * DV * FFN * 2);
    f16* Wvp1 = (f16*)alloc((size_t)3 * 3 * FSN * DV * 2);   // [kt][3*FS][D]
    f16* Wvp2 = (f16*)alloc((size_t)3 * 3 * FSN * FSN * 2);  // [kt][3*FS][FS]
    f16* Wmel = (f16*)alloc((size_t)NMELC * DV * 2);
    f16* Wpn1 = (f16*)alloc((size_t)5 * PNC * NMELC * 2);
    f16* Wpn2 = (f16*)alloc((size_t)5 * PNC * PNC * 2);
    f16* Wpn3 = (f16*)alloc((size_t)5 * PNC * PNC * 2);
    f16* Wpn4 = (f16*)alloc((size_t)5 * PNC * PNC * 2);
    f16* Wpn5 = (f16*)alloc((size_t)5 * NMELC * PNC * 2);
    float* xe   = (float*)alloc((size_t)BN_ * TN_ * DV * 4);
    float* xd   = (float*)alloc((size_t)BN_ * MLN * DV * 4);
    float* hb1  = (float*)alloc((size_t)BN_ * MLN * DV * 4);
    float* hb2  = (float*)alloc((size_t)BN_ * MLN * DV * 4);
    f16*   s16  = (f16*)  alloc((size_t)BN_ * MLN * FFN * 2);
    f16*   s16b = (f16*)  alloc((size_t)BN_ * MLN * FFN * 2);
    float* qkvb = (float*)alloc((size_t)BN_ * MLN * 3 * DV * 4);
    float* ob   = (float*)alloc((size_t)BN_ * MLN * DV * 4);
    float* ffb  = (float*)alloc((size_t)BN_ * MLN * FFN * 4);
    int*   cum  = (int*)  alloc((size_t)BN_ * TN_ * 4);
    float* t80  = (float*)alloc((size_t)BN_ * MLN * NMELC * 4);

    auto gsz = [](long n) {
        long b = (n + 255) / 256;
        if (b > (1L << 20)) b = (1L << 20);
        return dim3((unsigned)b);
    };
    auto gemm = [&](const f16* A, const f16* Bw, float* C, const float* bias,
                    f16* C16, int M, int N, int K, int flags, int seg, int sh) {
        dim3 grid((unsigned)((N + 127) / 128), (unsigned)((M + 127) / 128));
        gemm_f16_wmma<<<grid, 256, 0, stream>>>(A, Bw, C, bias, C16, M, N, K,
                                                flags, seg, sh);
    };
    auto cvt = [&](const float* in, f16* o2, long n) {
        cvt_f16_k<<<gsz(n), 256, 0, stream>>>(in, o2, n);
    };
    auto ln = [&](const float* in, float* o2, const float* g2, const float* b2,
                  int rows, int C2) {
        layernorm_k<<<rows, 256, 0, stream>>>(in, o2, g2, b2, C2);
    };

    // ---- one-time weight dequant to fp16 (~48 MB, stays L2-resident) ----
    dequant_k<<<gsz((long)LN_*4*DV*DV), 256, 0, stream>>>(e_ai, e_as, lut, WeA, (long)LN_*4*DV, DV);
    dequant_k<<<gsz((long)LN_*FFN*DV),  256, 0, stream>>>(e_ui, e_us, lut, WeU, (long)LN_*FFN, DV);
    dequant_k<<<gsz((long)LN_*DV*FFN),  256, 0, stream>>>(e_di, e_ds, lut, WeD, (long)LN_*DV, FFN);
    dequant_k<<<gsz((long)LN_*4*DV*DV), 256, 0, stream>>>(d_ai, d_as, lut, WdA, (long)LN_*4*DV, DV);
    dequant_k<<<gsz((long)LN_*FFN*DV),  256, 0, stream>>>(d_ui, d_us, lut, WdU, (long)LN_*FFN, DV);
    dequant_k<<<gsz((long)LN_*DV*FFN),  256, 0, stream>>>(d_di, d_ds, lut, WdD, (long)LN_*DV, FFN);
    taps_k<<<gsz((long)3*FSN*DV*3),   256, 0, stream>>>(vp_c1w, Wvp1, 3*FSN, DV, 3);
    taps_k<<<gsz((long)3*FSN*FSN*3),  256, 0, stream>>>(vp_c2w, Wvp2, 3*FSN, FSN, 3);
    cvt_f16_k<<<gsz((long)NMELC*DV),  256, 0, stream>>>(mel_w, Wmel, (long)NMELC*DV);
    taps_k<<<gsz((long)PNC*NMELC*5),  256, 0, stream>>>(pn_w1, Wpn1, PNC, NMELC, 5);
    taps_k<<<gsz((long)PNC*PNC*5),    256, 0, stream>>>(pn_w2, Wpn2, PNC, PNC, 5);
    taps_k<<<gsz((long)PNC*PNC*5),    256, 0, stream>>>(pn_w3, Wpn3, PNC, PNC, 5);
    taps_k<<<gsz((long)PNC*PNC*5),    256, 0, stream>>>(pn_w4, Wpn4, PNC, PNC, 5);
    taps_k<<<gsz((long)NMELC*PNC*5),  256, 0, stream>>>(pn_w5, Wpn5, NMELC, PNC, 5);

    // ---- transformer block (shared by encoder/decoder) ----
    auto layer = [&](float* x, int rows, int seq, const f16* Wat, const f16* Wup,
                     const f16* Wdn, const float* g2, const float* b2) {
        ln(x, hb1, g2, b2, rows, DV);
        cvt(hb1, s16, (long)rows * DV);
        // fused QKV: Wq/Wk/Wv rows are contiguous -> one GEMM, N = 3*D
        gemm(s16, Wat, qkvb, nullptr, nullptr, rows, 3 * DV, DV, 0, 0, 0);
        attention_k<<<BN_ * HN * seq, 256, 0, stream>>>(
            qkvb, qkvb + DV, qkvb + 2 * DV, ob, seq, 3 * DV, DV);
        cvt(ob, s16, (long)rows * DV);
        gemm(s16, Wat + 3 * (size_t)DV * DV, x, nullptr, nullptr,
             rows, DV, DV, EPI_ACCUM, 0, 0);
        ln(x, hb1, g2 + DV, b2 + DV, rows, DV);
        cvt(hb1, s16, (long)rows * DV);
        // up-proj + GELU, f16 mirror straight into s16b for the down-proj
        gemm(s16, Wup, ffb, nullptr, s16b, rows, FFN, DV, EPI_GELU, 0, 0);
        gemm(s16b, Wdn, x, nullptr, nullptr, rows, DV, FFN, EPI_ACCUM, 0, 0);
    };

    // ---- variance predictor (conv K=3 via 3 shifted accumulate-GEMMs) ----
    auto varpred = [&](const float* xin, int rows, int seq, int j, float* outp) {
        cvt(xin, s16, (long)rows * DV);
        for (int kt = 0; kt < 3; ++kt)
            gemm(s16, Wvp1 + ((size_t)kt * 3 * FSN + (size_t)j * FSN) * DV, hb1,
                 nullptr, nullptr, rows, FSN, DV, kt ? EPI_ACCUM : 0, seq, kt - 1);
        bias_relu_k<<<gsz((long)rows * FSN), 256, 0, stream>>>(hb1, vp_c1b + j * FSN,
                                                               (long)rows * FSN, FSN);
        ln(hb1, hb2, vp_g1 + j * FSN, vp_b1 + j * FSN, rows, FSN);
        cvt(hb2, s16, (long)rows * FSN);
        for (int kt = 0; kt < 3; ++kt)
            gemm(s16, Wvp2 + ((size_t)kt * 3 * FSN + (size_t)j * FSN) * FSN, hb1,
                 nullptr, nullptr, rows, FSN, FSN, kt ? EPI_ACCUM : 0, seq, kt - 1);
        bias_relu_k<<<gsz((long)rows * FSN), 256, 0, stream>>>(hb1, vp_c2b + j * FSN,
                                                               (long)rows * FSN, FSN);
        ln(hb1, hb2, vp_g2 + j * FSN, vp_b2 + j * FSN, rows, FSN);
        rowdot_k<<<rows, 256, 0, stream>>>(hb2, vp_lw + (size_t)j * FSN, vp_lb + j,
                                           outp, FSN);
    };

    // ================= pipeline =================
    embed_k<<<gsz((long)BN_ * TN_ * DV), 256, 0, stream>>>(text, emb, pos, xe,
                                                           TN_, DV, (long)BN_ * TN_ * DV);
    for (int l = 0; l < LN_; ++l)
        layer(xe, BN_ * TN_, TN_,
              WeA + (size_t)l * 4 * DV * DV, WeU + (size_t)l * FFN * DV,
              WeD + (size_t)l * DV * FFN,
              e_g + (size_t)l * 2 * DV, e_b + (size_t)l * 2 * DV);

    varpred(xe, BN_ * TN_, TN_, 0, o_dur);

    cumsum_k<<<BN_, 32, 0, stream>>>(durt, cum, TN_);
    lengthreg_k<<<BN_ * MLN, 256, 0, stream>>>(xe, cum, xd, TN_, MLN, DV);

    varpred(xd, BN_ * MLN, MLN, 1, o_pitch);
    addsc_k<<<gsz((long)BN_ * MLN * DV), 256, 0, stream>>>(xd, o_pitch, proj_w, proj_b,
                                                           (long)BN_ * MLN * DV, DV);
    varpred(xd, BN_ * MLN, MLN, 2, o_energy);
    addsc_k<<<gsz((long)BN_ * MLN * DV), 256, 0, stream>>>(xd, o_energy, proj_w + DV,
                                                           proj_b + DV,
                                                           (long)BN_ * MLN * DV, DV);
    addpos_k<<<gsz((long)BN_ * MLN * DV), 256, 0, stream>>>(xd, dpos,
                                                            (long)BN_ * MLN * DV, MLN, DV);

    for (int l = 0; l < LN_; ++l)
        layer(xd, BN_ * MLN, MLN,
              WdA + (size_t)l * 4 * DV * DV, WdU + (size_t)l * FFN * DV,
              WdD + (size_t)l * DV * FFN,
              d_g + (size_t)l * 2 * DV, d_b + (size_t)l * 2 * DV);

    // mel projection
    cvt(xd, s16, (long)BN_ * MLN * DV);
    gemm(s16, Wmel, o_mel, mel_b, nullptr, BN_ * MLN, NMELC, DV, EPI_BIAS, 0, 0);

    // ---- postnet: 5-tap convs as shifted accumulate-GEMMs ----
    cvt(o_mel, s16, (long)BN_ * MLN * NMELC);
    for (int kt = 0; kt < 5; ++kt)
        gemm(s16, Wpn1 + (size_t)kt * PNC * NMELC, ffb, nullptr, nullptr,
             BN_ * MLN, PNC, NMELC, kt ? EPI_ACCUM : 0, MLN, kt - 2);
    bn_tanh_k<<<gsz((long)BN_ * MLN * PNC), 256, 0, stream>>>(
        ffb, pn_b1, pn_g, pn_bb, s16, (long)BN_ * MLN * PNC, PNC);

    const f16*   Wpn[3]   = {Wpn2, Wpn3, Wpn4};
    const float* pbias[3] = {pn_b2, pn_b3, pn_b4};
    for (int jj = 0; jj < 3; ++jj) {
        for (int kt = 0; kt < 5; ++kt)
            gemm(s16, Wpn[jj] + (size_t)kt * PNC * PNC, ffb, nullptr, nullptr,
                 BN_ * MLN, PNC, PNC, kt ? EPI_ACCUM : 0, MLN, kt - 2);
        bn_tanh_k<<<gsz((long)BN_ * MLN * PNC), 256, 0, stream>>>(
            ffb, pbias[jj], pn_g + (size_t)(jj + 1) * PNC,
            pn_bb + (size_t)(jj + 1) * PNC, s16, (long)BN_ * MLN * PNC, PNC);
    }
    for (int kt = 0; kt < 5; ++kt)
        gemm(s16, Wpn5 + (size_t)kt * NMELC * PNC, t80, nullptr, nullptr,
             BN_ * MLN, NMELC, PNC, kt ? EPI_ACCUM : 0, MLN, kt - 2);
    postadd_k<<<gsz((long)BN_ * MLN * NMELC), 256, 0, stream>>>(
        o_mel, t80, pn_b5, o_post, (long)BN_ * MLN * NMELC, NMELC);
}
